// Int4Linear_11106785427742
// MI455X (gfx1250) — 8-bit matrix operand order repaired and checked by reading, compiled, not run on hardware
//
#include <hip/hip_runtime.h>
#include <hip/hip_bf16.h>
#include <stdint.h>


typedef __attribute__((ext_vector_type(8))) int       v8i;
typedef __attribute__((ext_vector_type(8))) _Float16  v8h;

#define BM 128
#define BN 128
#define BK 64
#define QMAX 7.0f
#define QEPS 1e-6f

__global__ __launch_bounds__(256)
void quant_rows_kernel(const _Float16* __restrict__ src,
                       int8_t* __restrict__ q,
                       float* __restrict__ scale,
                       int K) {
  const int row = blockIdx.x;
  const int tid = threadIdx.x;
  const size_t base = (size_t)row * (size_t)K;

  float amax = 0.0f;
  for (int k = tid * 8; k < K; k += 256 * 8) {
    v8h v = *(const v8h*)(src + base + k);
#pragma unroll
    for (int i = 0; i < 8; ++i) amax = fmaxf(amax, fabsf((float)v[i]));
  }
#pragma unroll
  for (int off = 16; off > 0; off >>= 1)
    amax = fmaxf(amax, __shfl_xor(amax, off, 32));

  __shared__ float wred[8];
  if ((tid & 31) == 0) wred[tid >> 5] = amax;
  __syncthreads();
  if (tid == 0) {
    float m = wred[0];
#pragma unroll
    for (int i = 1; i < 8; ++i) m = fmaxf(m, wred[i]);
    wred[0] = fmaxf(m, QEPS) / QMAX;
  }
  __syncthreads();
  const float s = wred[0];
  const float inv = 1.0f / s;

  for (int k = tid * 8; k < K; k += 256 * 8) {
    v8h v = *(const v8h*)(src + base + k);
    uint32_t lo = 0u, hi = 0u;
#pragma unroll
    for (int i = 0; i < 4; ++i) {
      int qi = (int)__builtin_rintf((float)v[i] * inv);
      qi = qi > 7 ? 7 : (qi < -7 ? -7 : qi);
      lo |= ((uint32_t)(uint8_t)(int8_t)qi) << (8 * i);
    }
#pragma unroll
    for (int i = 0; i < 4; ++i) {
      int qi = (int)__builtin_rintf((float)v[4 + i] * inv);
      qi = qi > 7 ? 7 : (qi < -7 ? -7 : qi);
      hi |= ((uint32_t)(uint8_t)(int8_t)qi) << (8 * i);
    }
    uint2 pk; pk.x = lo; pk.y = hi;
    *(uint2*)(q + base + k) = pk;
  }
  if (tid == 0) scale[row] = s;
}

__global__ __launch_bounds__(256)
void int4_wmma_gemm_kernel(const int8_t* __restrict__ qx,
                           const int8_t* __restrict__ qw,
                           const float* __restrict__ sx,
                           const float* __restrict__ sw,
                           const _Float16* __restrict__ bias,
                           _Float16* __restrict__ out,
                           int M, int N, int K) {
  __shared__ alignas(16) int8_t ldsA[2][BM * BK];
  __shared__ alignas(16) int8_t ldsB[2][BN * BK];

  const int tid  = threadIdx.x;
  const int wave = tid >> 5;
  const int lane = tid & 31;
  const int half = lane >> 4;
  const int l16  = lane & 15;

  const int mBase = blockIdx.y * BM;
  const int nBase = blockIdx.x * BN;

  const int wm = wave & 3;
  const int wn = wave >> 2;

  v8i acc[2][4];
  const v8i vzero = {0, 0, 0, 0, 0, 0, 0, 0};
#pragma unroll
  for (int mt = 0; mt < 2; ++mt)
#pragma unroll
    for (int nt = 0; nt < 4; ++nt) acc[mt][nt] = vzero;

  const int ldRow = tid >> 1;
  const int ldOff = (tid & 1) * 32;
  const int stage = ldRow * BK + ldOff;
  const int8_t* aPtr = qx + (size_t)(mBase + ldRow) * (size_t)K + ldOff;
  const int8_t* bPtr = qw + (size_t)(nBase + ldRow) * (size_t)K + ldOff;

  const uint32_t aLds0 = (uint32_t)(uintptr_t)&ldsA[0][stage];
  const uint32_t aLds1 = (uint32_t)(uintptr_t)&ldsA[1][stage];
  const uint32_t bLds0 = (uint32_t)(uintptr_t)&ldsB[0][stage];
  const uint32_t bLds1 = (uint32_t)(uintptr_t)&ldsB[1][stage];

  auto prefetch = [&](uint32_t aLds, uint32_t bLds, int kk) {
    const int8_t* ga = aPtr + kk;
    const int8_t* gb = bPtr + kk;
    asm volatile("global_load_async_to_lds_b128 %0, %1, off"
                 :: "v"(aLds), "v"(ga) : "memory");
    asm volatile("global_load_async_to_lds_b128 %0, %1, off offset:16"
                 :: "v"(aLds), "v"(ga) : "memory");
    asm volatile("global_load_async_to_lds_b128 %0, %1, off"
                 :: "v"(bLds), "v"(gb) : "memory");
    asm volatile("global_load_async_to_lds_b128 %0, %1, off offset:16"
                 :: "v"(bLds), "v"(gb) : "memory");
  };
  auto wait_async = [&]() {
    asm volatile("s_wait_asynccnt 0x0" ::: "memory");
  };

  auto compute = [&](const int8_t* As, const int8_t* Bs) {
    v8i afrag[2];
#pragma unroll
    for (int mt = 0; mt < 2; ++mt) {
      const int r = wm * 32 + mt * 16 + l16;
      const int8_t* p = As + r * BK + half * 8;
      uint2 t0 = *(const uint2*)(p + 0);
      uint2 t1 = *(const uint2*)(p + 16);
      uint2 t2 = *(const uint2*)(p + 32);
      uint2 t3 = *(const uint2*)(p + 48);
      v8i a;
      a[0] = (int)t0.x; a[1] = (int)t0.y;
      a[2] = (int)t1.x; a[3] = (int)t1.y;
      a[4] = (int)t2.x; a[5] = (int)t2.y;
      a[6] = (int)t3.x; a[7] = (int)t3.y;
      afrag[mt] = a;
    }
    v8i bfrag[4];
#pragma unroll
    for (int nt = 0; nt < 4; ++nt) {
      const int r = wn * 64 + nt * 16 + l16;
      const int8_t* p = Bs + r * BK + half * 8;
      uint2 u0 = *(const uint2*)(p + 0);
      uint2 u1 = *(const uint2*)(p + 16);
      uint2 u2 = *(const uint2*)(p + 32);
      uint2 u3 = *(const uint2*)(p + 48);
      v8i b;
      b[0] = (int)u0.x; b[1] = (int)u0.y; b[2] = (int)u1.x; b[3] = (int)u1.y;
      b[4] = (int)u2.x; b[5] = (int)u2.y; b[6] = (int)u3.x; b[7] = (int)u3.y;
      bfrag[nt] = b;
    }
#pragma unroll
    for (int mt = 0; mt < 2; ++mt)
#pragma unroll
      for (int nt = 0; nt < 4; ++nt)
        acc[mt][nt] = __builtin_amdgcn_wmma_i32_16x16x64_iu8(
             true, afrag[mt],  true, bfrag[nt],
            acc[mt][nt],  false,  false);
  };

  prefetch(aLds0, bLds0, 0);
  for (int kk = 0; kk < K; kk += 2 * BK) {
    wait_async();
    __syncthreads();
    if (kk + BK < K) prefetch(aLds1, bLds1, kk + BK);
    compute(&ldsA[0][0], &ldsB[0][0]);

    wait_async();
    __syncthreads();
    if (kk + 2 * BK < K) prefetch(aLds0, bLds0, kk + 2 * BK);
    compute(&ldsA[1][0], &ldsB[1][0]);
  }

#pragma unroll
  for (int mt = 0; mt < 2; ++mt) {
#pragma unroll
    for (int nt = 0; nt < 4; ++nt) {
      const int rowBase = mBase + wm * 32 + mt * 16 + half * 8;
      const int col     = nBase + wn * 64 + nt * 16 + l16;
      const float     swv = sw[col];
      const _Float16  bv  = bias[col];
#pragma unroll
      for (int j = 0; j < 8; ++j) {
        const int row = rowBase + j;
        const float f = (float)acc[mt][nt][j] * sx[row] * swv;
        out[(size_t)row * (size_t)N + col] = (_Float16)f + bv;
      }
    }
  }
}

extern "C" void kernel_launch(void* const* d_in, const int* in_sizes, int n_in,
                              void* d_out, int out_size, void* d_ws, size_t ws_size,
                              hipStream_t stream) {
  const _Float16* x    = (const _Float16*)d_in[0];
  const _Float16* w    = (const _Float16*)d_in[1];
  const _Float16* bias = (const _Float16*)d_in[2];

  const int N = in_sizes[2];
  const int K = in_sizes[1] / N;
  const int M = in_sizes[0] / K;

  _Float16* out = (_Float16*)d_out;

  int8_t* q_x = (int8_t*)d_ws;
  int8_t* q_w = q_x + (size_t)M * (size_t)K;
  float*  s_x = (float*)(q_w + (size_t)N * (size_t)K);
  float*  s_w = s_x + M;

  quant_rows_kernel<<<M, 256, 0, stream>>>(x, q_x, s_x, K);
  quant_rows_kernel<<<N, 256, 0, stream>>>(w, q_w, s_w, K);

  dim3 grid(N / BN, M / BM);
  int4_wmma_gemm_kernel<<<grid, 256, 0, stream>>>(q_x, q_w, s_x, s_w, bias,
                                                  out, M, N, K);
}
